// HeterogeneousGNNLayer_11209864642593
// MI455X (gfx1250) — matrix operand order repaired and checked by reading, not compiled, not run on hardware
//
#include <hip/hip_runtime.h>

// Problem constants (match reference)
#define NL 16384
#define NG 32768
#define ND 8192
#define DD 256
#define HH 4

typedef __attribute__((ext_vector_type(16))) _Float16 v16h;
typedef __attribute__((ext_vector_type(8)))  _Float16 v8h;
typedef __attribute__((ext_vector_type(4)))  _Float16 v4h;
typedef __attribute__((ext_vector_type(8)))  float    v8f;
typedef __attribute__((ext_vector_type(4)))  float    v4f;

union AV { v16h v; v8h h[2]; };
union HV { v8h v; _Float16 e[8]; };

// ---------------- f32 -> f16 conversion (vectorized x4) ----------------
__global__ void __launch_bounds__(256) cvt_f32_f16(const float* __restrict__ in,
                                                   _Float16* __restrict__ out, int n4) {
  int t = blockIdx.x * blockDim.x + threadIdx.x;
  if (t >= n4) return;
  v4f x = *(const v4f*)(in + (size_t)t * 4);
  v4h y;
  y[0] = (_Float16)x[0]; y[1] = (_Float16)x[1];
  y[2] = (_Float16)x[2]; y[3] = (_Float16)x[3];
  *(v4h*)(out + (size_t)t * 4) = y;
}

// ---------------- zero fill (vectorized x4 floats) ----------------
__global__ void __launch_bounds__(256) zero_f32(float* __restrict__ p, long n4) {
  long t = (long)blockIdx.x * blockDim.x + threadIdx.x;
  if (t >= n4) return;
  v4f z = {0.f, 0.f, 0.f, 0.f};
  *(v4f*)(p + t * 4) = z;
}

// ---------------- WMMA GEMM: Y[N,256] = X[N,256] @ W^T (all f16, f32 acc) ---------
// One wave per 16x16 output tile. Lane data layouts per CDNA5 ISA 7.12.2:
//   A 16x32 f16: lanes 0-15 hold M=lane, K = kb+{0..7, 16..23};
//                lanes 16-31 hold M=lane-16, K = kb+{8..15, 24..31}.
//   B 32x16 f16: lane holds N=lane&15, same K order as A (kb+8*(lane>>4)+{0..7, 16..23}).
//   B[k][n] = W[n][k] (row-major W) -> two contiguous 16B chunks per lane.
__global__ void __launch_bounds__(256) gemm_nt_wmma(const _Float16* __restrict__ X,
                                                    const _Float16* __restrict__ W,
                                                    _Float16* __restrict__ Y) {
  const int wave = (int)((blockIdx.x * blockDim.x + threadIdx.x) >> 5);
  const int lane = threadIdx.x & 31;
  const int mt = wave >> 4;     // row tile (grid sized so Nrows = total waves)
  const int nt = wave & 15;     // col tile, D=256 -> 16 tiles
  const int l15 = lane & 15;
  const int hi  = lane >> 4;
  const size_t rowA = (size_t)(mt * 16 + l15) * DD;
  const size_t rowB = (size_t)(nt * 16 + l15) * DD;
  const int hiA = hi * 8;
  const int hiB = hi * 16;
  v8f acc = {};
#pragma unroll
  for (int kb = 0; kb < DD; kb += 32) {
    AV a, b;
    a.h[0] = *(const v8h*)(X + rowA + kb + hiA);        // K = kb+hiA .. +7
    a.h[1] = *(const v8h*)(X + rowA + kb + hiA + 16);   // K = kb+hiA+16 .. +23
    b.h[0] = *(const v8h*)(W + rowB + kb + hiA);        // K = kb+hiA .. +7
    b.h[1] = *(const v8h*)(W + rowB + kb + hiA + 16);   // K = kb+hiA+16 .. +23
    acc = __builtin_amdgcn_wmma_f32_16x16x32_f16(false, a.v, false, b.v,
                                                 (short)0, acc, false, false);
  }
  // D layout: VGPR r -> (M = r + 8*(lane>=16), N = lane&15)
  const int col = nt * 16 + l15;
  const int rbase = mt * 16 + hi * 8;
#pragma unroll
  for (int r = 0; r < 8; r++) {
    Y[(size_t)(rbase + r) * DD + col] = (_Float16)acc[r];
  }
}

// ---------------- edge scatter: msg[dst] += w * Xw[src], deg[dst] += w ------------
// One wave per edge: edge index is wave-uniform -> src/dst/w become scalar (SMEM)
// loads via readfirstlane; each lane covers 8 contiguous columns (one 16B load).
__global__ void __launch_bounds__(256) scatter_edges(const _Float16* __restrict__ Xw,
                                                     const int* __restrict__ src,
                                                     const int* __restrict__ dst,
                                                     const float* __restrict__ w,
                                                     float* __restrict__ msg,
                                                     float* __restrict__ deg, int nE) {
  int wave = (int)((blockIdx.x * blockDim.x + threadIdx.x) >> 5);
  int lane = threadIdx.x & 31;
  if (wave >= nE) return;
  const int e = __builtin_amdgcn_readfirstlane(wave);   // force scalar edge index
  const int s  = src[e];
  const int d  = dst[e];
  const float we = w[e];
  const int c = lane * 8;
  HV x;
  x.v = *(const v8h*)(Xw + (size_t)s * DD + c);
  float* mp = msg + (size_t)d * DD + c;
#pragma unroll
  for (int k = 0; k < 8; k++) {
    atomicAdd(mp + k, we * (float)x.e[k]);
  }
  if (lane == 0) atomicAdd(deg + d, we);
}

// ---------------- per-node update: attention + self + residual + LN + leaky -------
// One wave per node, lane covers columns lane+32j, j=0..7.
__global__ void __launch_bounds__(256) update_nodes(const float* __restrict__ msg0,
                                                    const float* __restrict__ msg1,
                                                    const float* __restrict__ deg0,
                                                    const float* __restrict__ deg1,
                                                    const _Float16* __restrict__ Sf,
                                                    const float* __restrict__ h,
                                                    const float* __restrict__ A0,
                                                    const float* __restrict__ A1,
                                                    const float* __restrict__ gamma,
                                                    const float* __restrict__ beta,
                                                    float* __restrict__ out, int N) {
  int node = (int)((blockIdx.x * blockDim.x + threadIdx.x) >> 5);
  int lane = threadIdx.x & 31;
  if (node >= N) return;
  const size_t base = (size_t)node * DD;
  float d0 = deg0[node]; d0 = (d0 == 0.f) ? 1.f : d0;
  float d1 = deg1[node]; d1 = (d1 == 0.f) ? 1.f : d1;
  const float i0 = 1.f / d0, i1 = 1.f / d1;
  float m0[8], m1[8];
  float s0[HH] = {0.f, 0.f, 0.f, 0.f}, s1[HH] = {0.f, 0.f, 0.f, 0.f};
#pragma unroll
  for (int j = 0; j < 8; j++) {
    int c = lane + j * 32;
    m0[j] = msg0[base + c] * i0;
    m1[j] = msg1[base + c] * i1;
#pragma unroll
    for (int hh = 0; hh < HH; hh++) {
      s0[hh] = fmaf(m0[j], A0[hh * DD + c], s0[hh]);
      s1[hh] = fmaf(m1[j], A1[hh * DD + c], s1[hh]);
    }
  }
#pragma unroll
  for (int off = 16; off > 0; off >>= 1) {
#pragma unroll
    for (int hh = 0; hh < HH; hh++) {
      s0[hh] += __shfl_xor(s0[hh], off, 32);
      s1[hh] += __shfl_xor(s1[hh], off, 32);
    }
  }
  // scores/(sqrt(D)*TEMP) = score * 1/(16*0.5) = score * 0.125
  const float scl = 0.125f;
  float c0 = 0.f, c1 = 0.f;
#pragma unroll
  for (int hh = 0; hh < HH; hh++) {
    float a = s0[hh] * scl, b = s1[hh] * scl;
    float mx = fmaxf(a, b);
    float e0 = __expf(a - mx), e1 = __expf(b - mx);
    float inv = 1.f / (e0 + e1);
    c0 += e0 * inv; c1 += e1 * inv;
  }
  c0 *= 0.25f; c1 *= 0.25f;   // mean over H=4 heads
  float u[8], sum = 0.f, ss = 0.f;
#pragma unroll
  for (int j = 0; j < 8; j++) {
    int c = lane + j * 32;
    float v = (float)Sf[base + c] + c0 * m0[j] + c1 * m1[j] + h[base + c];
    u[j] = v; sum += v; ss += v * v;
  }
#pragma unroll
  for (int off = 16; off > 0; off >>= 1) {
    sum += __shfl_xor(sum, off, 32);
    ss  += __shfl_xor(ss, off, 32);
  }
  const float mean = sum * (1.f / DD);
  const float var  = ss * (1.f / DD) - mean * mean;
  const float rstd = rsqrtf(var + 1e-5f);
#pragma unroll
  for (int j = 0; j < 8; j++) {
    int c = lane + j * 32;
    float v = (u[j] - mean) * rstd * gamma[c] + beta[c];
    out[base + c] = (v > 0.f) ? v : 0.01f * v;
  }
}

// =============================== launcher =======================================
extern "C" void kernel_launch(void* const* d_in, const int* in_sizes, int n_in,
                              void* d_out, int out_size, void* d_ws, size_t ws_size,
                              hipStream_t stream) {
  const float* h_l = (const float*)d_in[0];
  const float* h_g = (const float*)d_in[1];
  const float* h_d = (const float*)d_in[2];
  // input order per setup_inputs(): relations l2g, g2l, g2d, d2g, l2d, d2l
  struct Rel { const int* src; const int* dst; const float* w; const float* W;
               const float* A; int E; };
  Rel rel[6];
  for (int r = 0; r < 6; r++) {
    rel[r].src = (const int*)d_in[3 + r * 5 + 0];
    rel[r].dst = (const int*)d_in[3 + r * 5 + 1];
    rel[r].w   = (const float*)d_in[3 + r * 5 + 2];
    rel[r].W   = (const float*)d_in[3 + r * 5 + 3];
    rel[r].A   = (const float*)d_in[3 + r * 5 + 4];
    rel[r].E   = in_sizes[3 + r * 5 + 0];
  }
  const float* Wself[3] = {(const float*)d_in[33], (const float*)d_in[36], (const float*)d_in[39]};
  const float* gam[3]   = {(const float*)d_in[34], (const float*)d_in[37], (const float*)d_in[40]};
  const float* bet[3]   = {(const float*)d_in[35], (const float*)d_in[38], (const float*)d_in[41]};

  // ---- workspace carve-out (256B aligned slabs) ----
  char* ws = (char*)d_ws;
  size_t off = 0;
  auto alloc = [&](size_t bytes) -> void* {
    void* p = ws + off;
    off = (off + bytes + 255) & ~(size_t)255;
    return p;
  };
  _Float16* hf_l = (_Float16*)alloc((size_t)NL * DD * 2);
  _Float16* hf_g = (_Float16*)alloc((size_t)NG * DD * 2);
  _Float16* hf_d = (_Float16*)alloc((size_t)ND * DD * 2);
  _Float16* Wf[9];
  for (int i = 0; i < 9; i++) Wf[i] = (_Float16*)alloc((size_t)DD * DD * 2);
  _Float16* Xw   = (_Float16*)alloc((size_t)NG * DD * 2);   // reused per relation
  _Float16* Sf_l = (_Float16*)alloc((size_t)NL * DD * 2);
  _Float16* Sf_g = (_Float16*)alloc((size_t)NG * DD * 2);
  _Float16* Sf_d = (_Float16*)alloc((size_t)ND * DD * 2);
  size_t zeroStart = off;
  float* msg_g2l = (float*)alloc((size_t)NL * DD * 4);
  float* msg_d2l = (float*)alloc((size_t)NL * DD * 4);
  float* msg_l2g = (float*)alloc((size_t)NG * DD * 4);
  float* msg_d2g = (float*)alloc((size_t)NG * DD * 4);
  float* msg_g2d = (float*)alloc((size_t)ND * DD * 4);
  float* msg_l2d = (float*)alloc((size_t)ND * DD * 4);
  float* deg_g2l = (float*)alloc((size_t)NL * 4);
  float* deg_d2l = (float*)alloc((size_t)NL * 4);
  float* deg_l2g = (float*)alloc((size_t)NG * 4);
  float* deg_d2g = (float*)alloc((size_t)NG * 4);
  float* deg_g2d = (float*)alloc((size_t)ND * 4);
  float* deg_l2d = (float*)alloc((size_t)ND * 4);
  size_t zeroEnd = off;

  // ---- 1) f16 conversions ----
  auto cvt = [&](const float* in, _Float16* o, long n) {
    int n4 = (int)(n >> 2);
    cvt_f32_f16<<<dim3((n4 + 255) / 256), dim3(256), 0, stream>>>(in, o, n4);
  };
  cvt(h_l, hf_l, (long)NL * DD);
  cvt(h_g, hf_g, (long)NG * DD);
  cvt(h_d, hf_d, (long)ND * DD);
  for (int r = 0; r < 6; r++) cvt(rel[r].W, Wf[r], (long)DD * DD);
  for (int t = 0; t < 3; t++) cvt(Wself[t], Wf[6 + t], (long)DD * DD);

  // ---- 2) zero msg + deg (every launch: atomics accumulate) ----
  {
    long n4 = (long)((zeroEnd - zeroStart) / 16);
    zero_f32<<<dim3((unsigned)((n4 + 255) / 256)), dim3(256), 0, stream>>>(
        (float*)(ws + zeroStart), n4);
  }

  // ---- 3) per relation: hoisted WMMA transform, then edge scatter ----
  auto gemm = [&](const _Float16* X, const _Float16* Wm, _Float16* Y, int Nrows) {
    gemm_nt_wmma<<<dim3(Nrows / 8), dim3(256), 0, stream>>>(X, Wm, Y);
  };
  struct Job { int r; const _Float16* hf; int ns; float* msg; float* deg; };
  Job jobs[6] = {
    {0, hf_l, NL, msg_l2g, deg_l2g},   // l2g
    {1, hf_g, NG, msg_g2l, deg_g2l},   // g2l
    {2, hf_g, NG, msg_g2d, deg_g2d},   // g2d
    {3, hf_d, ND, msg_d2g, deg_d2g},   // d2g
    {4, hf_l, NL, msg_l2d, deg_l2d},   // l2d
    {5, hf_d, ND, msg_d2l, deg_d2l},   // d2l
  };
  for (int j = 0; j < 6; j++) {
    const Job& J = jobs[j];
    gemm(J.hf, Wf[J.r], Xw, J.ns);
    long tthreads = (long)rel[J.r].E * 32;   // one wave per edge
    scatter_edges<<<dim3((unsigned)((tthreads + 255) / 256)), dim3(256), 0, stream>>>(
        Xw, rel[J.r].src, rel[J.r].dst, rel[J.r].w, J.msg, J.deg, rel[J.r].E);
  }

  // ---- 4) self transforms ----
  gemm(hf_l, Wf[6], Sf_l, NL);
  gemm(hf_g, Wf[7], Sf_g, NG);
  gemm(hf_d, Wf[8], Sf_d, ND);

  // ---- 5) node updates ----
  float* out  = (float*)d_out;
  float* outl = out;
  float* outg = out + (size_t)NL * DD;
  float* outd = out + (size_t)(NL + NG) * DD;
  // out_l: msgs [m_g2l, m_d2l], atts [A_g2l(rel1), A_d2l(rel5)]
  update_nodes<<<dim3(NL / 8), dim3(256), 0, stream>>>(
      msg_g2l, msg_d2l, deg_g2l, deg_d2l, Sf_l, h_l,
      rel[1].A, rel[5].A, gam[0], bet[0], outl, NL);
  // out_g: msgs [m_l2g, m_d2g], atts [A_l2g(rel0), A_d2g(rel3)]
  update_nodes<<<dim3(NG / 8), dim3(256), 0, stream>>>(
      msg_l2g, msg_d2g, deg_l2g, deg_d2g, Sf_g, h_g,
      rel[0].A, rel[3].A, gam[1], bet[1], outg, NG);
  // out_d: msgs [m_g2d, m_l2d], atts [A_g2d(rel2), A_l2d(rel4)]
  update_nodes<<<dim3(ND / 8), dim3(256), 0, stream>>>(
      msg_g2d, msg_l2d, deg_g2d, deg_l2d, Sf_d, h_d,
      rel[2].A, rel[4].A, gam[2], bet[2], outd, ND);
  (void)n_in; (void)out_size; (void)ws_size;
}
